// MultiHeadAttention_4355096838770
// MI455X (gfx1250) — matrix operand order repaired and checked by reading, not compiled, not run on hardware
//
#include <hip/hip_runtime.h>

// MI455X / gfx1250, wave32. All matmuls via v_wmma_f32_16x16x32_bf16.
typedef __bf16 bf16;
typedef __attribute__((ext_vector_type(16))) __bf16 v16bf;
typedef __attribute__((ext_vector_type(8)))  __bf16 bf16x8;
typedef __attribute__((ext_vector_type(4)))  __bf16 bf16x4;
typedef __attribute__((ext_vector_type(8)))  float   v8f;

#define LDSTR 40  // padded LDS row stride (bf16): 80B rows -> conflict-free b128 lane reads

union AFrag { v16bf v; bf16x8 h[2]; };

static __device__ __forceinline__ v8f wmma_bf16(v16bf a, v16bf b, v8f c) {
  // (neg_a, A, neg_b, B, c_mod, C, reuse_a, reuse_b)
  return __builtin_amdgcn_wmma_f32_16x16x32_bf16(false, a, false, b, (short)0, c, false, false);
}

// ---------------------------------------------------------------------------
// Generic tiled GEMM:  Out[M,N] = A[M,K] * B[K,N] + bias[N]
//   A_F32:   A is f32 (convert->bf16 in LDS), else A already bf16
//   OUT_TRANS: store Out transposed (Out[n*ldo + m], bf16) -- used for V
//   OUT_F32:   store f32 (final projection to d_out), else bf16
// Block tile 128x128, K-step 32. 8 waves; each wave computes 32x64.
// ---------------------------------------------------------------------------
template <bool A_F32, bool OUT_TRANS, bool OUT_F32>
__global__ __launch_bounds__(256) void gemm_wmma_kernel(
    const void* __restrict__ Aptr, int lda,
    const float* __restrict__ Bp, int ldb,
    const float* __restrict__ bias,
    void* __restrict__ Out, int ldo, int K) {
  __shared__ bf16 sA[128 * LDSTR];   // [m][k], K-innermost
  __shared__ bf16 sBt[128 * LDSTR];  // [n][k], K-innermost (B micro-transposed)

  const int tid = threadIdx.x;
  const int wave = tid >> 5, lane = tid & 31;
  const int half = lane >> 4, l15 = lane & 15;
  const int wm = wave & 3, wn = wave >> 2;
  const int mb0 = blockIdx.y * 128, nb0 = blockIdx.x * 128;

  v8f acc[2][4] = {};

  const int arow = tid >> 1;        // A staging: 2 threads/row
  const int acol = (tid & 1) * 16;
  const int kq = (tid & 7) * 4;     // B staging: 4x4 micro-transpose
  const int nq = (tid >> 3) * 4;

  for (int kb = 0; kb < K; kb += 32) {
    __syncthreads();
    // ---- stage A tile (128x32) as bf16, K-innermost ----
    if (A_F32) {
      const float* a = (const float*)Aptr + (size_t)(mb0 + arow) * lda + kb + acol;
      float4 f0 = ((const float4*)a)[0];
      float4 f1 = ((const float4*)a)[1];
      float4 f2 = ((const float4*)a)[2];
      float4 f3 = ((const float4*)a)[3];
      bf16x8 h0 = {(bf16)f0.x, (bf16)f0.y, (bf16)f0.z, (bf16)f0.w,
                   (bf16)f1.x, (bf16)f1.y, (bf16)f1.z, (bf16)f1.w};
      bf16x8 h1 = {(bf16)f2.x, (bf16)f2.y, (bf16)f2.z, (bf16)f2.w,
                   (bf16)f3.x, (bf16)f3.y, (bf16)f3.z, (bf16)f3.w};
      *(bf16x8*)&sA[arow * LDSTR + acol] = h0;
      *(bf16x8*)&sA[arow * LDSTR + acol + 8] = h1;
    } else {
      const bf16* a = (const bf16*)Aptr + (size_t)(mb0 + arow) * lda + kb + acol;
      *(bf16x8*)&sA[arow * LDSTR + acol] = ((const bf16x8*)a)[0];
      *(bf16x8*)&sA[arow * LDSTR + acol + 8] = ((const bf16x8*)a)[1];
    }
    // ---- stage B tile (32x128) -> sBt[n][k] via 4x4 register transpose ----
    {
      const float* bp = Bp + (size_t)(kb + kq) * ldb + nb0 + nq;
      float4 r0 = *(const float4*)bp;
      float4 r1 = *(const float4*)(bp + (size_t)ldb);
      float4 r2 = *(const float4*)(bp + 2 * (size_t)ldb);
      float4 r3 = *(const float4*)(bp + 3 * (size_t)ldb);
      bf16x4 c0 = {(bf16)r0.x, (bf16)r1.x, (bf16)r2.x, (bf16)r3.x};
      bf16x4 c1 = {(bf16)r0.y, (bf16)r1.y, (bf16)r2.y, (bf16)r3.y};
      bf16x4 c2 = {(bf16)r0.z, (bf16)r1.z, (bf16)r2.z, (bf16)r3.z};
      bf16x4 c3 = {(bf16)r0.w, (bf16)r1.w, (bf16)r2.w, (bf16)r3.w};
      *(bf16x4*)&sBt[(nq + 0) * LDSTR + kq] = c0;
      *(bf16x4*)&sBt[(nq + 1) * LDSTR + kq] = c1;
      *(bf16x4*)&sBt[(nq + 2) * LDSTR + kq] = c2;
      *(bf16x4*)&sBt[(nq + 3) * LDSTR + kq] = c3;
    }
    if (kb + 32 < K)
      __builtin_prefetch(Bp + (size_t)(kb + 32) * ldb + nb0 + nq, 0, 1);
    __syncthreads();
    // ---- fragments + WMMA ----
    AFrag af[2], bfr[4];
#pragma unroll
    for (int t = 0; t < 2; ++t) {
      int r = wm * 32 + t * 16 + l15;
      af[t].h[0] = *(const bf16x8*)&sA[r * LDSTR + 8 * half];
      af[t].h[1] = *(const bf16x8*)&sA[r * LDSTR + 16 + 8 * half];
    }
#pragma unroll
    for (int u = 0; u < 4; ++u) {
      int n = wn * 64 + u * 16 + l15;
      bfr[u].h[0] = *(const bf16x8*)&sBt[n * LDSTR + 8 * half];
      bfr[u].h[1] = *(const bf16x8*)&sBt[n * LDSTR + 16 + 8 * half];
    }
#pragma unroll
    for (int t = 0; t < 2; ++t)
#pragma unroll
      for (int u = 0; u < 4; ++u)
        acc[t][u] = wmma_bf16(af[t].v, bfr[u].v, acc[t][u]);
  }

  // ---- epilogue ----
#pragma unroll
  for (int t = 0; t < 2; ++t) {
#pragma unroll
    for (int u = 0; u < 4; ++u) {
      int m0 = mb0 + wm * 32 + t * 16 + 8 * half;  // rows m0..m0+7 in this frag
      int n = nb0 + wn * 64 + u * 16 + l15;
      float bv = bias[n];
      if (OUT_TRANS) {
        bf16x8 p;
#pragma unroll
        for (int r = 0; r < 8; ++r) p[r] = (bf16)(acc[t][u][r] + bv);
        *(bf16x8*)((bf16*)Out + (size_t)n * ldo + m0) = p;  // 16B store
      } else if (OUT_F32) {
        float* o = (float*)Out;
#pragma unroll
        for (int r = 0; r < 8; ++r)
          o[(size_t)(m0 + r) * ldo + n] = acc[t][u][r] + bv;
      } else {
        bf16* o = (bf16*)Out;
#pragma unroll
        for (int r = 0; r < 8; ++r)
          o[(size_t)(m0 + r) * ldo + n] = (bf16)(acc[t][u][r] + bv);
      }
    }
  }
}

// ---------------------------------------------------------------------------
// Logits: per (b,h)  L = (Q Kt)/sqrt(512) + masks, f32 -> attn (H,B,S,S).
// Q and K are both K-innermost (head dim contiguous) -> no transpose needed.
// WG: 128 q-rows x 64 k-cols; wave = 16x64 strip (4 tiles).
// ---------------------------------------------------------------------------
__global__ __launch_bounds__(256) void attn_logits_kernel(
    const bf16* __restrict__ Qp, const bf16* __restrict__ Kp,
    const int* __restrict__ pad, const int* __restrict__ la,
    float* __restrict__ attn) {
  __shared__ bf16 sK[64 * LDSTR];
  const int tid = threadIdx.x, wave = tid >> 5, lane = tid & 31;
  const int half = lane >> 4, l15 = lane & 15;
  const int b = blockIdx.z & 3, h = blockIdx.z >> 2;
  const int kcb = blockIdx.x * 64;
  const int qr0 = blockIdx.y * 128 + wave * 16;

  const bf16* qrow = Qp + (size_t)(b * 1024 + qr0 + l15) * 4096 + h * 512;
  const int srow = tid >> 2, scol = (tid & 3) * 8;
  const bf16* krow = Kp + (size_t)(b * 1024 + kcb + srow) * 4096 + h * 512;

  v8f acc[4] = {};
  for (int kb = 0; kb < 512; kb += 32) {
    __syncthreads();
    *(bf16x8*)&sK[srow * LDSTR + scol] = *(const bf16x8*)(krow + kb + scol);
    __syncthreads();
    AFrag a;
    a.h[0] = *(const bf16x8*)(qrow + kb + 8 * half);
    a.h[1] = *(const bf16x8*)(qrow + kb + 16 + 8 * half);
#pragma unroll
    for (int u = 0; u < 4; ++u) {
      AFrag kf;
      int j = u * 16 + l15;
      kf.h[0] = *(const bf16x8*)&sK[j * LDSTR + 8 * half];
      kf.h[1] = *(const bf16x8*)&sK[j * LDSTR + 16 + 8 * half];
      acc[u] = wmma_bf16(a.v, kf.v, acc[u]);
    }
  }
  const float SCALE = 0.04419417382415922f;  // 1/sqrt(512)
#pragma unroll
  for (int u = 0; u < 4; ++u) {
    int kc = kcb + u * 16 + l15;
#pragma unroll
    for (int r = 0; r < 8; ++r) {
      int qr = qr0 + 8 * half + r;
      size_t mrow = (size_t)(b * 1024 + qr) * 1024 + kc;  // (B,S,S) mask index
      float v = acc[u][r] * SCALE;
      if (pad[mrow]) v -= 1e15f;
      if (la[mrow]) v -= 1e15f;
      attn[((size_t)(h * 4 + b) * 1024 + qr) * 1024 + kc] = v;  // (H,B,S,S)
    }
  }
}

// ---------------------------------------------------------------------------
// In-place row softmax over attn (H*B*S rows of 1024).
// ---------------------------------------------------------------------------
__global__ __launch_bounds__(256) void softmax_kernel(float* __restrict__ attn) {
  __shared__ float red[256];
  float* p = attn + (size_t)blockIdx.x * 1024;
  const int t = threadIdx.x;
  float v0 = p[t], v1 = p[t + 256], v2 = p[t + 512], v3 = p[t + 768];
  float mx = fmaxf(fmaxf(v0, v1), fmaxf(v2, v3));
  red[t] = mx;
  __syncthreads();
  for (int s = 128; s > 0; s >>= 1) {
    if (t < s) red[t] = fmaxf(red[t], red[t + s]);
    __syncthreads();
  }
  mx = red[0];
  __syncthreads();
  v0 = __expf(v0 - mx); v1 = __expf(v1 - mx);
  v2 = __expf(v2 - mx); v3 = __expf(v3 - mx);
  red[t] = v0 + v1 + v2 + v3;
  __syncthreads();
  for (int s = 128; s > 0; s >>= 1) {
    if (t < s) red[t] += red[t + s];
    __syncthreads();
  }
  float inv = 1.0f / red[0];
  p[t] = v0 * inv; p[t + 256] = v1 * inv;
  p[t + 512] = v2 * inv; p[t + 768] = v3 * inv;
}

// ---------------------------------------------------------------------------
// Ctx = attn @ V per (b,h). attn f32 rows -> bf16 fragments in registers;
// Vt stored transposed (d-major, s-innermost) so B fragments are contiguous.
// ---------------------------------------------------------------------------
__global__ __launch_bounds__(256) void attn_v_kernel(
    const float* __restrict__ attn, const bf16* __restrict__ Vt,
    bf16* __restrict__ Ctx) {
  __shared__ bf16 sV[64 * LDSTR];
  const int tid = threadIdx.x, wave = tid >> 5, lane = tid & 31;
  const int half = lane >> 4, l15 = lane & 15;
  const int b = blockIdx.z & 3, h = blockIdx.z >> 2;
  const int nb = blockIdx.x * 64;
  const int qr0 = blockIdx.y * 128 + wave * 16;

  const float* arow = attn + ((size_t)(h * 4 + b) * 1024 + qr0 + l15) * 1024;
  const int srow = tid >> 2, scol = (tid & 3) * 8;
  const bf16* vrow = Vt + (size_t)(h * 512 + nb + srow) * 4096 + b * 1024;

  v8f acc[4] = {};
  for (int kb = 0; kb < 1024; kb += 32) {
    __syncthreads();
    *(bf16x8*)&sV[srow * LDSTR + scol] = *(const bf16x8*)(vrow + kb + scol);
    __syncthreads();
    AFrag a;
    {
      const float* p0 = arow + kb + 8 * half;
      float4 f0 = *(const float4*)p0;
      float4 f1 = *(const float4*)(p0 + 4);
      const float* p1 = arow + kb + 16 + 8 * half;
      float4 f2 = *(const float4*)p1;
      float4 f3 = *(const float4*)(p1 + 4);
      bf16x8 t0 = {(bf16)f0.x, (bf16)f0.y, (bf16)f0.z, (bf16)f0.w,
                   (bf16)f1.x, (bf16)f1.y, (bf16)f1.z, (bf16)f1.w};
      bf16x8 t1 = {(bf16)f2.x, (bf16)f2.y, (bf16)f2.z, (bf16)f2.w,
                   (bf16)f3.x, (bf16)f3.y, (bf16)f3.z, (bf16)f3.w};
      a.h[0] = t0;
      a.h[1] = t1;
    }
#pragma unroll
    for (int u = 0; u < 4; ++u) {
      AFrag vf;
      int j = u * 16 + l15;
      vf.h[0] = *(const bf16x8*)&sV[j * LDSTR + 8 * half];
      vf.h[1] = *(const bf16x8*)&sV[j * LDSTR + 16 + 8 * half];
      acc[u] = wmma_bf16(a.v, vf.v, acc[u]);
    }
  }
#pragma unroll
  for (int u = 0; u < 4; ++u) {
    int d = nb + u * 16 + l15;
#pragma unroll
    for (int r = 0; r < 8; ++r) {
      int qr = qr0 + 8 * half + r;
      Ctx[(size_t)(b * 1024 + qr) * 4096 + h * 512 + d] = (bf16)acc[u][r];
    }
  }
}

// ---------------------------------------------------------------------------
extern "C" void kernel_launch(void* const* d_in, const int* in_sizes, int n_in,
                              void* d_out, int out_size, void* d_ws, size_t ws_size,
                              hipStream_t stream) {
  (void)in_sizes; (void)n_in; (void)out_size; (void)ws_size;
  const float* query = (const float*)d_in[0];
  const float* keys = (const float*)d_in[1];
  const float* vals = (const float*)d_in[2];
  const int* pad = (const int*)d_in[3];
  const int* la = (const int*)d_in[4];
  const float* Wq = (const float*)d_in[5];
  const float* bq = (const float*)d_in[6];
  const float* Wk = (const float*)d_in[7];
  const float* bk = (const float*)d_in[8];
  const float* Wv = (const float*)d_in[9];
  const float* bv = (const float*)d_in[10];
  const float* Wo = (const float*)d_in[11];
  const float* bo = (const float*)d_in[12];

  float* out = (float*)d_out;
  float* attn = out + (size_t)4 * 1024 * 512;  // (H,B,S,S) region

  bf16* Qp = (bf16*)d_ws;                     // (B*S, H*512) bf16
  bf16* Kp = Qp + (size_t)4096 * 4096;        // (B*S, H*512) bf16
  bf16* Vt = Kp + (size_t)4096 * 4096;        // transposed: (H*512, B*S) bf16
  bf16* Ctx = Vt + (size_t)4096 * 4096;       // (B*S, H*512) bf16

  dim3 blk(256);
  dim3 gP(32, 32);  // 4096x4096 output in 128x128 tiles
  gemm_wmma_kernel<true, false, false><<<gP, blk, 0, stream>>>(query, 512, Wq, 4096, bq, Qp, 4096, 512);
  gemm_wmma_kernel<true, false, false><<<gP, blk, 0, stream>>>(keys, 512, Wk, 4096, bk, Kp, 4096, 512);
  gemm_wmma_kernel<true, true, false><<<gP, blk, 0, stream>>>(vals, 512, Wv, 4096, bv, Vt, 4096, 512);

  attn_logits_kernel<<<dim3(16, 8, 32), blk, 0, stream>>>(Qp, Kp, pad, la, attn);
  softmax_kernel<<<dim3(32768), blk, 0, stream>>>(attn);
  attn_v_kernel<<<dim3(8, 8, 32), blk, 0, stream>>>(attn, Vt, Ctx);

  gemm_wmma_kernel<false, false, true><<<dim3(4, 32), blk, 0, stream>>>(Ctx, 4096, Wo, 512, bo, out, 512, 4096);
}
